// Ec_63900523430195
// MI455X (gfx1250) — matrix operand order repaired and checked by reading, not compiled, not run on hardware
//
#include <hip/hip_runtime.h>
#include <hip/hip_bf16.h>

typedef unsigned int   u32;
typedef unsigned short u16;

typedef __attribute__((ext_vector_type(16))) __bf16 bf16x16;
typedef __attribute__((ext_vector_type(8)))  float  v8f;

union FragAB { uint4 q[2]; bf16x16 v; };

#define TT   192
#define CHID 512
#define NB   1024

__device__ __forceinline__ u16 f2bf(float f) {
  u32 u = __float_as_uint(f);
  u32 r = u + 0x7FFFu + ((u >> 16) & 1u);   // round-to-nearest-even
  return (u16)(r >> 16);
}

__device__ __forceinline__ u32 hash3(u32 a, u32 b, u32 c) {
  u32 h = a * 0x8DA6B343u + b * 0xD8163841u + c * 0xCB1AB31Fu + 0x9E3779B9u;
  h ^= h >> 16; h *= 0x7FEB352Du;
  h ^= h >> 15; h *= 0x846CA68Bu;
  h ^= h >> 16;
  return h;
}

// ---------------------------------------------------------------------------
// Weight packing: f32 (Cout, Cin, K) -> bf16 [k][co][ci_pad], zero padded K-dim
// ---------------------------------------------------------------------------
__global__ void pack_conv_k(const float* __restrict__ w, u16* __restrict__ wp,
                            int Cout, int Cin, int CinPad, int K) {
  int idx = blockIdx.x * blockDim.x + threadIdx.x;
  int total = K * Cout * CinPad;
  if (idx >= total) return;
  int k  = idx / (Cout * CinPad);
  int r  = idx - k * (Cout * CinPad);
  int co = r / CinPad;
  int ci = r - co * CinPad;
  float v = (ci < Cin) ? w[((size_t)co * Cin + ci) * K + k] : 0.0f;
  wp[idx] = f2bf(v);
}

__global__ void pack_lstm_k(const float* __restrict__ wf, const float* __restrict__ wb,
                            const float* __restrict__ bif, const float* __restrict__ bhf,
                            const float* __restrict__ bib, const float* __restrict__ bhb,
                            u16* __restrict__ wp, float* __restrict__ bl) {
  int idx = blockIdx.x * blockDim.x + threadIdx.x;
  if (idx < 64 * CHID) {
    int g = idx / CHID, ci = idx - g * CHID;
    float v = (g < 32) ? wf[g * CHID + ci] : wb[(g - 32) * CHID + ci];
    wp[idx] = f2bf(v);
  }
  if (idx < 64)
    bl[idx] = (idx < 32) ? (bif[idx] + bhf[idx]) : (bib[idx - 32] + bhb[idx - 32]);
}

// ---------------------------------------------------------------------------
// Implicit-GEMM conv1d ('same', KTAPS taps) via v_wmma_f32_16x16x32_bf16.
// All shape parameters are compile-time so the K loop fully unrolls into a
// dense run of WMMAs, staging index math is strength-reduced, and the weight
// row for the next tap is prefetched (global_prefetch_b8) while the current
// tap's WMMAs execute.
// Block: 128 thr (4 waves). Each wave owns a 16(co)x16(t) tile; waves share a
// bf16 LDS slab of the input: [16+2*pad rows of t] x CINPAD (<=20KB).
// Grid: (T/16, COUT/64, B).
// ---------------------------------------------------------------------------
template <int CIN, int CINPAD, int KTAPS, int COUT, int RELU>
__global__ __launch_bounds__(128) void conv_wmma(
    const float* __restrict__ X, const u16* __restrict__ Wp,
    const float* __restrict__ bias, float* __restrict__ Y) {
  constexpr int PAD = KTAPS >> 1;
  constexpr int STW = 16 + 2 * PAD;
  __shared__ alignas(16) u16 sx[STW * CINPAD];

  const int b    = blockIdx.z;
  const int t0   = blockIdx.x * 16;
  const int co0  = blockIdx.y * 64;
  const int tid  = threadIdx.x;
  const int wave = tid >> 5;
  const int lane = tid & 31;
  const int half = lane >> 4;
  const int l16  = lane & 15;

  // Stage input slab as bf16 into LDS: sx[tt][ci], tt = t - (t0-PAD)
  constexpr int TOTAL = STW * CINPAD;
  for (int idx = tid; idx < TOTAL; idx += 128) {
    int ci = idx / STW;                 // compile-time STW -> magic multiply
    int tt = idx - ci * STW;            // consecutive threads -> consecutive t
    int t  = t0 - PAD + tt;
    float v = 0.0f;
    if (ci < CIN && t >= 0 && t < TT)
      v = X[((size_t)b * CIN + ci) * TT + t];
    sx[tt * CINPAD + ci] = f2bf(v);
  }
  __syncthreads();

  v8f acc = {0.f, 0.f, 0.f, 0.f, 0.f, 0.f, 0.f, 0.f};
  const int co = co0 + wave * 16 + l16;      // A-frag row owned by this lane
  const u16* wbase = Wp + (size_t)co * CINPAD;

  for (int k = 0; k < KTAPS; ++k) {
    const u16* wrow = wbase + (size_t)k * COUT * CINPAD;  // A row (M=co), K=ci
    const u16* brow = sx + (l16 + k) * CINPAD;            // B col (N=t),  K=ci
    if (k + 1 < KTAPS)                                    // hide L2 latency
      __builtin_prefetch(wbase + (size_t)(k + 1) * COUT * CINPAD, 0, 1);
    #pragma unroll
    for (int kk = 0; kk < CINPAD; kk += 32) {
      FragAB a, bb;
      // A 16x32 bf16: lanes 0-15 hold K = 0..7,16..23; lanes 16-31: 8..15,24..31
      a.q[0]  = *(const uint4*)(wrow + kk + half * 8);
      a.q[1]  = *(const uint4*)(wrow + kk + 16 + half * 8);
      // B 32x16 bf16: lanes 0-15 hold K = 0..7,16..23; lanes 16-31: 8..15,24..31
      bb.q[0] = *(const uint4*)(brow + kk + half * 8);
      bb.q[1] = *(const uint4*)(brow + kk + 16 + half * 8);
      acc = __builtin_amdgcn_wmma_f32_16x16x32_bf16(
              false, a.v, false, bb.v, (short)0, acc, false, false);
    }
  }

  // D layout: vgpr v -> row M = v + 8*half; lane l16 -> column N = t
  #pragma unroll
  for (int v = 0; v < 8; ++v) {
    int cor = co0 + wave * 16 + v + 8 * half;
    float val = acc[v] + bias[cor];
    if (RELU) val = fmaxf(val, 0.0f);
    Y[((size_t)b * COUT + cor) * TT + t0 + l16] = val;
  }
}

// ---------------------------------------------------------------------------
// InterpLnr (training path, len_seq == 192). One 512-thread block per batch.
// Deterministic counter-hash PRNG stands in for jax threefry.
// Operates directly in (B, C, T) layout.
// ---------------------------------------------------------------------------
__global__ __launch_bounds__(512) void interp_lnr_k(
    const float* __restrict__ X, float* __restrict__ Y, int keyid) {
  __shared__ float s_scale[7];
  __shared__ int   s_len[7];
  __shared__ float s_lam[448];
  __shared__ int   s_src[448];
  __shared__ int   s_dst[448];
  __shared__ int   s_scan[512];

  const int b   = blockIdx.x;
  const int tid = threadIdx.x;

  if (tid < 7) {
    u32 n = (u32)(b * 7 + tid);
    float u01 = (float)(hash3((u32)keyid, n, 0u) >> 8) * (1.0f / 16777216.0f);
    s_scale[tid] = 0.5f + u01;                                   // [0.5, 1.5)
    s_len[tid]   = 19 + (int)(hash3((u32)keyid, n, 1u) % 13u);   // [19, 32)
  }
  __syncthreads();

  int m = 0;
  if (tid < 448) {
    int s = tid >> 6;            // segment 0..6
    int i = tid & 63;            // position within 2*MAX_SEG
    float isc = (float)i / s_scale[s];
    float fl  = floorf(isc);
    s_lam[tid] = isc - fl;
    int len = s_len[s];
    int offs = 0;
    for (int s2 = 0; s2 < s; ++s2) offs += s_len[s2];   // exclusive cumsum
    int idx_org = (int)fl + offs;
    m = (fl < (float)(len - 1)) && (idx_org < TT - 1);
    s_src[tid] = m ? idx_org : 0;
  }
  s_scan[tid] = (tid < 448) ? m : 0;
  __syncthreads();
  for (int off = 1; off < 512; off <<= 1) {   // inclusive Hillis-Steele scan
    int v = (tid >= off) ? s_scan[tid - off] : 0;
    __syncthreads();
    s_scan[tid] += v;
    __syncthreads();
  }
  if (tid < 448) {
    int pos = s_scan[tid] - 1;
    s_dst[tid] = (m && pos < TT) ? pos : -1;   // pos >= 192 -> dumped
  }
  __syncthreads();

  // Phase 2: tid == channel. Zero then packed scatter (dests unique, increasing).
  const float* xr = X + ((size_t)b * CHID + tid) * TT;
  float*       yr = Y + ((size_t)b * CHID + tid) * TT;
  for (int t = 0; t < TT; ++t) yr[t] = 0.0f;
  for (int j = 0; j < 448; ++j) {
    int d = s_dst[j];
    if (d >= 0) {
      int f  = s_src[j];
      int cl = (f + 1 < TT) ? f + 1 : TT - 1;
      float lam = s_lam[j];
      yr[d] = (1.0f - lam) * xr[f] + lam * xr[cl];
    }
  }
}

// ---------------------------------------------------------------------------
// GroupNorm(32 groups over 512ch) + ReLU, in place; one block per (b, group).
// ---------------------------------------------------------------------------
__global__ __launch_bounds__(256) void gn_relu_k(
    float* __restrict__ X, const float* __restrict__ W, const float* __restrict__ Bs) {
  __shared__ float s0[256], s1[256];
  const int g   = blockIdx.x & 31;
  const int b   = blockIdx.x >> 5;
  const int tid = threadIdx.x;
  float* base = X + ((size_t)b * CHID + g * 16) * TT;
  const int N = 16 * TT;
  float sum = 0.f, sq = 0.f;
  for (int i = tid; i < N; i += 256) { float v = base[i]; sum += v; sq += v * v; }
  s0[tid] = sum; s1[tid] = sq;
  __syncthreads();
  for (int off = 128; off > 0; off >>= 1) {
    if (tid < off) { s0[tid] += s0[tid + off]; s1[tid] += s1[tid + off]; }
    __syncthreads();
  }
  float mu  = s0[0] / (float)N;
  float var = s1[0] / (float)N - mu * mu;
  float rs  = rsqrtf(var + 1e-5f);
  for (int i = tid; i < N; i += 256) {
    int ch = g * 16 + i / TT;
    float v = (base[i] - mu) * rs * W[ch] + Bs[ch];
    base[i] = fmaxf(v, 0.0f);
  }
}

// ---------------------------------------------------------------------------
// Bi-LSTM recurrence (H=8). One wave per (batch, direction): 32 lanes =
// 4 gates x 8 units; hidden state broadcast via wave32 __shfl.
// ---------------------------------------------------------------------------
__global__ __launch_bounds__(64) void lstm_k(
    const float* __restrict__ G, const float* __restrict__ whh_f,
    const float* __restrict__ whh_b, float* __restrict__ out) {
  const int b    = blockIdx.x;
  const int dir  = threadIdx.x >> 5;
  const int lane = threadIdx.x & 31;
  const int u    = lane & 7;
  const float* whh = dir ? whh_b : whh_f;
  float wr[8];
  #pragma unroll
  for (int j = 0; j < 8; ++j) wr[j] = whh[lane * 8 + j];
  const float* grow = G + ((size_t)b * 64 + dir * 32 + lane) * TT;
  float h = 0.f, c = 0.f;
  for (int step = 0; step < TT; ++step) {
    int t = dir ? (TT - 1 - step) : step;
    float gv = grow[t];
    #pragma unroll
    for (int j = 0; j < 8; ++j) gv += wr[j] * __shfl(h, j);
    float iv = __shfl(gv, u);
    float fv = __shfl(gv, 8 + u);
    float gg = __shfl(gv, 16 + u);
    float ov = __shfl(gv, 24 + u);
    float si = 1.0f / (1.0f + __expf(-iv));
    float sf = 1.0f / (1.0f + __expf(-fv));
    float so = 1.0f / (1.0f + __expf(-ov));
    c = sf * c + si * tanhf(gg);
    h = so * tanhf(c);
    if (lane < 8) {
      if (dir == 0) {
        if ((t & 7) == 7) out[((size_t)b * 24 + (t >> 3)) * 16 + u] = h;
      } else {
        if ((t & 7) == 0) out[((size_t)b * 24 + (t >> 3)) * 16 + 8 + u] = h;
      }
    }
  }
}

// ---------------------------------------------------------------------------
// Orchestration. Workspace (~862 MB):
//   act0 (402.7MB) | act1 (402.7MB) | G (50.3MB) | wp1/wp2/wp3/wpl bf16 | bl
// ---------------------------------------------------------------------------
extern "C" void kernel_launch(void* const* d_in, const int* in_sizes, int n_in,
                              void* d_out, int out_size, void* d_ws, size_t ws_size,
                              hipStream_t stream) {
  (void)in_sizes; (void)n_in; (void)out_size; (void)ws_size;
  const float* c    = (const float*)d_in[0];
  const float* w1   = (const float*)d_in[1];
  const float* b1   = (const float*)d_in[2];
  const float* w2   = (const float*)d_in[3];
  const float* b2   = (const float*)d_in[4];
  const float* w3   = (const float*)d_in[5];
  const float* b3   = (const float*)d_in[6];
  const float* gnw  = (const float*)d_in[7];
  const float* gnb  = (const float*)d_in[8];
  const float* wihf = (const float*)d_in[9];
  const float* whhf = (const float*)d_in[10];
  const float* bihf = (const float*)d_in[11];
  const float* bhhf = (const float*)d_in[12];
  const float* wihb = (const float*)d_in[13];
  const float* whhb = (const float*)d_in[14];
  const float* bihb = (const float*)d_in[15];
  const float* bhhb = (const float*)d_in[16];

  char* ws = (char*)d_ws;
  const size_t S_ACT = (size_t)NB * CHID * TT * sizeof(float);
  const size_t S_G   = (size_t)NB * 64 * TT * sizeof(float);
  float* act0 = (float*)(ws);
  float* act1 = (float*)(ws + S_ACT);
  float* Gbuf = (float*)(ws + 2 * S_ACT);
  u16*   wp1  = (u16*)(ws + 2 * S_ACT + S_G);
  u16*   wp2  = (u16*)((char*)wp1 + (size_t)5 * 512 * 96 * 2);
  u16*   wp3  = (u16*)((char*)wp2 + (size_t)5 * 512 * 512 * 2);
  u16*   wpl  = (u16*)((char*)wp3 + (size_t)5 * 512 * 512 * 2);
  float* bl   = (float*)((char*)wpl + (size_t)64 * 512 * 2);

  { int tot = 5 * 512 * 96;  pack_conv_k<<<(tot + 255) / 256, 256, 0, stream>>>(w1, wp1, 512, 80, 96, 5); }
  { int tot = 5 * 512 * 512; pack_conv_k<<<(tot + 255) / 256, 256, 0, stream>>>(w2, wp2, 512, 512, 512, 5); }
  { int tot = 5 * 512 * 512; pack_conv_k<<<(tot + 255) / 256, 256, 0, stream>>>(w3, wp3, 512, 512, 512, 5); }
  pack_lstm_k<<<(64 * CHID + 255) / 256, 256, 0, stream>>>(wihf, wihb, bihf, bhhf, bihb, bhhb, wpl, bl);

  dim3 gconv(TT / 16, CHID / 64, NB);
  conv_wmma<80, 96, 5, 512, 1><<<gconv, 128, 0, stream>>>(c,    wp1, b1, act0);
  interp_lnr_k<<<NB, 512, 0, stream>>>(act0, act1, 0);
  conv_wmma<512, 512, 5, 512, 1><<<gconv, 128, 0, stream>>>(act1, wp2, b2, act0);
  interp_lnr_k<<<NB, 512, 0, stream>>>(act0, act1, 1);
  conv_wmma<512, 512, 5, 512, 1><<<gconv, 128, 0, stream>>>(act1, wp3, b3, act0);
  interp_lnr_k<<<NB, 512, 0, stream>>>(act0, act1, 2);
  gn_relu_k<<<NB * 32, 256, 0, stream>>>(act1, gnw, gnb);
  interp_lnr_k<<<NB, 512, 0, stream>>>(act1, act0, 3);
  dim3 glstm(TT / 16, 1, NB);
  conv_wmma<512, 512, 1, 64, 0><<<glstm, 128, 0, stream>>>(act0, wpl, bl, Gbuf);
  lstm_k<<<NB, 64, 0, stream>>>(Gbuf, whhf, whhb, (float*)d_out);
}
